// MMIL_Net_90168543412583
// MI455X (gfx1250) — compile-verified, matrix operand order checked by reading, not run on hardware
//
#include <hip/hip_runtime.h>
#include <hip/hip_bf16.h>

typedef __attribute__((ext_vector_type(16))) __bf16 bf16x16;
typedef __attribute__((ext_vector_type(8)))  __bf16 bf16x8;
typedef __attribute__((ext_vector_type(8)))  float  f32x8;

#define DEVI static __device__ __forceinline__

DEVI float siluf(float x) { return x / (1.f + __expf(-x)); }

// ---------------------------------------------------------------------------
// GEMM:  C(M,N) = act(A(M,K) @ W(N,K)^T + bias)
// A: bf16 row-major (lda), W: bf16 row-major (N,K)
// C: optional f32 out (ldc), Cbf: optional bf16 out (ldc)
// One wave computes a 16x64 tile: one A fragment feeds 4 WMMAs per K-step.
// Requires M%16==0, N%64==0, K%32==0. act: 0=none, 1=relu, 2=softplus
// ---------------------------------------------------------------------------
__global__ void gemm_bf16_wmma(const __bf16* __restrict__ A, int lda,
                               const __bf16* __restrict__ W,
                               const float* __restrict__ bias,
                               float* __restrict__ C,
                               __bf16* __restrict__ Cbf, int ldc,
                               int M, int N, int K, int act)
{
    const int wave    = threadIdx.x >> 5;
    const int lane    = threadIdx.x & 31;
    const int tilesN4 = N >> 6;                      // 64-wide column groups
    const int total   = (M >> 4) * tilesN4;
    const int tileId  = blockIdx.x * 8 + wave;
    if (tileId >= total) return;                     // wave-uniform: EXEC stays all-1s
    const int tileM  = tileId / tilesN4;
    const int tileN4 = tileId % tilesN4;

    const int half = lane >> 4;                      // K sub-band select (+0 / +8)
    const int ml   = lane & 15;                      // M row (A) / N col (B)

    const __bf16* arow = A + (size_t)(tileM * 16 + ml) * lda + half * 8;
    const __bf16* wrow = W + (size_t)(tileN4 * 64 + ml) * K  + half * 8;
    const size_t  wstep = (size_t)16 * K;            // next 16-column group

    f32x8 acc[4] = {{}, {}, {}, {}};

    for (int k0 = 0; k0 < K; k0 += 32) {
        // A fragment: elements 0..7 -> K=k0+half*8.., 8..15 -> K=k0+16+half*8..
        const bf16x8 a0 = *(const bf16x8*)(arow + k0);
        const bf16x8 a1 = *(const bf16x8*)(arow + k0 + 16);
        bf16x16 av;
        #pragma unroll
        for (int e = 0; e < 8; ++e) { av[e] = a0[e]; av[e + 8] = a1[e]; }

        __builtin_prefetch((const void*)(arow + k0 + 32), 0, 1);

        #pragma unroll
        for (int t = 0; t < 4; ++t) {
            const __bf16* wp = wrow + (size_t)t * wstep + k0;
            const bf16x8 w0 = *(const bf16x8*)(wp);
            const bf16x8 w1 = *(const bf16x8*)(wp + 16);
            bf16x16 bv;
            #pragma unroll
            for (int e = 0; e < 8; ++e) { bv[e] = w0[e]; bv[e + 8] = w1[e]; }
            // 8 args: (neg_a, A, neg_b, B, c_mod, C, reuse_a, reuse_b)
            acc[t] = __builtin_amdgcn_wmma_f32_16x16x32_bf16(false, av, false, bv,
                                                             (short)0, acc[t], false, false);
        }
    }

    #pragma unroll
    for (int t = 0; t < 4; ++t) {
        const int col = tileN4 * 64 + t * 16 + ml;   // C/D layout: lane%16 = N
        const float bz = bias ? bias[col] : 0.f;
        #pragma unroll
        for (int r = 0; r < 8; ++r) {                // VGPR r = M row (+8 upper half)
            const int row = tileM * 16 + half * 8 + r;
            float v = acc[t][r] + bz;
            if (act == 1)      v = fmaxf(v, 0.f);
            else if (act == 2) v = (v > 20.f) ? v : log1pf(__expf(v));
            if (C)   C[(size_t)row * ldc + col]   = v;
            if (Cbf) Cbf[(size_t)row * ldc + col] = (__bf16)v;
        }
    }
}

// ---------------------------------------------------------------------------
__global__ void cvt_f32_bf16(const float* __restrict__ src, __bf16* __restrict__ dst, int n)
{
    int i = blockIdx.x * 256 + threadIdx.x;
    if (i < n) dst[i] = (__bf16)src[i];
}

// avm[b,i,j,d] = (i<10&&j<10)? q[b,i,d]*v[b,j,d] : 0  + (j<10? q[b,j,d]:0) + (i<10? v[b,i,d]:0)
// written directly as bf16 (GEMM operand)
__global__ void build_avm(const float* __restrict__ q, const float* __restrict__ v,
                          __bf16* __restrict__ avm)
{
    int idx = blockIdx.x * 256 + threadIdx.x;        // 64*11*11*512 elements
    int d = idx & 511;
    int t = idx >> 9;
    int j = t % 11; t /= 11;
    int i = t % 11;
    int b = t / 11;
    float s = 0.f;
    if (i < 10) {
        float vi = v[((size_t)b * 10 + i) * 512 + d];
        s += vi;
        if (j < 10) s += q[((size_t)b * 10 + i) * 512 + d] * v[((size_t)b * 10 + j) * 512 + d];
    }
    if (j < 10) s += q[((size_t)b * 10 + j) * 512 + d];
    avm[idx] = (__bf16)s;
}

// causal depthwise conv (K=4) over L + bias + SiLU; x = xz[..., :1024]
// writes f32 (for scan) and bf16 (for x_proj GEMM)
__global__ void conv_silu(const float* __restrict__ xz, const float* __restrict__ cw,
                          const float* __restrict__ cb, float* __restrict__ xc,
                          __bf16* __restrict__ xcb)
{
    int idx = blockIdx.x * 256 + threadIdx.x;        // 704*11*1024
    int c = idx & 1023;
    int t = idx >> 10;
    int l = t % 11;
    int n = t / 11;
    float s = cb[c];
    #pragma unroll
    for (int k = 0; k < 4; ++k) {
        int j = l + k - 3;
        if (j >= 0) s += cw[c * 4 + k] * xz[((size_t)n * 11 + j) * 2048 + c];
    }
    const float r = siluf(s);
    xc[idx]  = r;
    xcb[idx] = (__bf16)r;
}

// selective scan: one block per (sequence, channel-chunk); h state in registers
__global__ void mamba_scan(const float* __restrict__ xz, const float* __restrict__ xc,
                           const float* __restrict__ dbl, const float* __restrict__ delta,
                           const float* __restrict__ A_log, const float* __restrict__ D_p,
                           __bf16* __restrict__ yout)
{
    __shared__ float sB[11 * 16];
    __shared__ float sC[11 * 16];
    const int n = blockIdx.x;
    for (int i = threadIdx.x; i < 11 * 16; i += 256) {
        int l = i / 16, s = i % 16;
        const float* dp = dbl + ((size_t)n * 11 + l) * 64;
        sB[i] = dp[32 + s];
        sC[i] = dp[48 + s];
    }
    __syncthreads();

    const int d = blockIdx.y * 256 + threadIdx.x;    // 0..1023
    float Av[16];
    #pragma unroll
    for (int s = 0; s < 16; ++s) Av[s] = -__expf(A_log[(size_t)d * 16 + s]);
    const float Dv = D_p[d];

    float h[16];
    #pragma unroll
    for (int s = 0; s < 16; ++s) h[s] = 0.f;

    for (int l = 0; l < 11; ++l) {
        const size_t r = (size_t)n * 11 + l;
        const float dt = delta[r * 1024 + d];
        const float xt = xc[r * 1024 + d];
        const float dx = dt * xt;
        float y = 0.f;
        #pragma unroll
        for (int s = 0; s < 16; ++s) {
            const float dA = __expf(dt * Av[s]);
            h[s] = h[s] * dA + dx * sB[l * 16 + s];
            y += h[s] * sC[l * 16 + s];
        }
        y += xt * Dv;
        const float z = xz[r * 2048 + 1024 + d];
        yout[r * 1024 + d] = (__bf16)(y * siluf(z));
    }
}

// compressed/audio_features: af[m] = proj_b + sum_l F[m*11+l]*proj_w[l]
__global__ void compress_proj(const float* __restrict__ F, const float* __restrict__ pw,
                              const float* __restrict__ pb, float* __restrict__ af, int n)
{
    int m = blockIdx.x * 256 + threadIdx.x;
    if (m >= n) return;
    const float* p = F + (size_t)m * 11;
    float s = pb[0];
    #pragma unroll
    for (int l = 0; l < 11; ++l) s += p[l] * pw[l];
    af[m] = s;
}

DEVI void block_reduce(float* red, int tid)
{
    __syncthreads();
    for (int s = 128; s > 0; s >>= 1) {
        if (tid < s) red[tid] += red[tid + s];
        __syncthreads();
    }
}

// q = LN1(src_q + 0.5*(af[:, :-1] + af[:, 1:]))   one block per row, D=512
// writes f32 (final residual) and bf16 (lin1 GEMM operand)
__global__ void src1_ln1(const float* __restrict__ src_q, const float* __restrict__ af,
                         const float* __restrict__ g, const float* __restrict__ b,
                         float* __restrict__ q, __bf16* __restrict__ qb)
{
    __shared__ float red[256];
    const int row = blockIdx.x;                      // 0..639
    const int bi = row / 10, t = row % 10;
    const int tid = threadIdx.x;
    float v[2];
    #pragma unroll
    for (int e = 0; e < 2; ++e) {
        const int d = tid + e * 256;
        const float a0 = af[((size_t)bi * 11 + t) * 512 + d];
        const float a1 = af[((size_t)bi * 11 + t + 1) * 512 + d];
        v[e] = src_q[(size_t)row * 512 + d] + 0.5f * (a0 + a1);
    }
    red[tid] = v[0] + v[1];
    block_reduce(red, tid);
    const float mu = red[0] * (1.f / 512.f);
    __syncthreads();
    const float d0 = v[0] - mu, d1 = v[1] - mu;
    red[tid] = d0 * d0 + d1 * d1;
    block_reduce(red, tid);
    const float rs = rsqrtf(red[0] * (1.f / 512.f) + 1e-5f);
    #pragma unroll
    for (int e = 0; e < 2; ++e) {
        const int d = tid + e * 256;
        const float o = (v[e] - mu) * rs * g[d] + b[d];
        q[(size_t)row * 512 + d]  = o;
        qb[(size_t)row * 512 + d] = (__bf16)o;
    }
}

// out = LN2(q + ff)
__global__ void ln2_final(const float* __restrict__ q, const float* __restrict__ ff,
                          const float* __restrict__ g, const float* __restrict__ b,
                          float* __restrict__ out)
{
    __shared__ float red[256];
    const int row = blockIdx.x;
    const int tid = threadIdx.x;
    float v[2];
    #pragma unroll
    for (int e = 0; e < 2; ++e) {
        const int d = tid + e * 256;
        v[e] = q[(size_t)row * 512 + d] + ff[(size_t)row * 512 + d];
    }
    red[tid] = v[0] + v[1];
    block_reduce(red, tid);
    const float mu = red[0] * (1.f / 512.f);
    __syncthreads();
    const float d0 = v[0] - mu, d1 = v[1] - mu;
    red[tid] = d0 * d0 + d1 * d1;
    block_reduce(red, tid);
    const float rs = rsqrtf(red[0] * (1.f / 512.f) + 1e-5f);
    #pragma unroll
    for (int e = 0; e < 2; ++e) {
        const int d = tid + e * 256;
        out[(size_t)row * 512 + d] = (v[e] - mu) * rs * g[d] + b[d];
    }
}

// ---------------------------------------------------------------------------
extern "C" void kernel_launch(void* const* d_in, const int* in_sizes, int n_in,
                              void* d_out, int out_size, void* d_ws, size_t ws_size,
                              hipStream_t stream)
{
    (void)in_sizes; (void)n_in; (void)out_size; (void)ws_size;

    const float* src_q     = (const float*)d_in[0];
    const float* src_v     = (const float*)d_in[1];
    const float* in_proj_w = (const float*)d_in[2];
    const float* conv_w    = (const float*)d_in[3];
    const float* conv_b    = (const float*)d_in[4];
    const float* x_proj_w  = (const float*)d_in[5];
    const float* dt_proj_w = (const float*)d_in[6];
    const float* dt_proj_b = (const float*)d_in[7];
    const float* A_log     = (const float*)d_in[8];
    const float* D_p       = (const float*)d_in[9];
    const float* out_proj_w= (const float*)d_in[10];
    const float* proj_w    = (const float*)d_in[11];
    const float* proj_b    = (const float*)d_in[12];
    const float* ln1_g     = (const float*)d_in[13];
    const float* ln1_b     = (const float*)d_in[14];
    const float* lin1_w    = (const float*)d_in[15];
    const float* lin1_b    = (const float*)d_in[16];
    const float* lin2_w    = (const float*)d_in[17];
    const float* lin2_b    = (const float*)d_in[18];
    const float* ln2_g     = (const float*)d_in[19];
    const float* ln2_b     = (const float*)d_in[20];
    float* out = (float*)d_out;

    // workspace carve-out (256B aligned)
    char* ws = (char*)d_ws;
    size_t off = 0;
    auto take = [&](size_t bytes) -> char* {
        char* p = ws + off;
        off = (off + bytes + 255) & ~(size_t)255;
        return p;
    };
    const int MR = 7744;                             // 704 sequences * L=11
    __bf16* avm_b = (__bf16*)take((size_t)MR * 512 * 2);
    float*  xz    = (float*)take((size_t)MR * 2048 * 4);
    float*  xc    = (float*)take((size_t)MR * 1024 * 4);
    __bf16* xc_b  = (__bf16*)take((size_t)MR * 1024 * 2);
    float*  dbl   = (float*)take((size_t)MR * 64 * 4);
    __bf16* dbl_b = (__bf16*)take((size_t)MR * 64 * 2);
    float*  delta = (float*)take((size_t)MR * 1024 * 4);
    __bf16* yact_b= (__bf16*)take((size_t)MR * 1024 * 2);
    float*  oaud  = (float*)take((size_t)MR * 512 * 4);
    float*  af    = (float*)take((size_t)360448 * 4);
    float*  qbuf  = (float*)take((size_t)640 * 512 * 4);
    __bf16* q_b   = (__bf16*)take((size_t)640 * 512 * 2);
    __bf16* h1_b  = (__bf16*)take((size_t)640 * 512 * 2);
    float*  ffb   = (float*)take((size_t)640 * 512 * 4);
    __bf16* w_in  = (__bf16*)take((size_t)2048 * 512 * 2);
    __bf16* w_xp  = (__bf16*)take((size_t)64 * 1024 * 2);
    __bf16* w_dt  = (__bf16*)take((size_t)1024 * 32 * 2);
    __bf16* w_op  = (__bf16*)take((size_t)512 * 1024 * 2);
    __bf16* w_l1  = (__bf16*)take((size_t)512 * 512 * 2);
    __bf16* w_l2  = (__bf16*)take((size_t)512 * 512 * 2);

    auto cvt = [&](const float* s, __bf16* dst, int n) {
        cvt_f32_bf16<<<(n + 255) / 256, 256, 0, stream>>>(s, dst, n);
    };
    auto gemm = [&](const __bf16* A, int lda, const __bf16* W, const float* bias,
                    float* C, __bf16* Cbf, int ldc, int M, int N, int K, int act) {
        const int tiles = (M / 16) * (N / 64);       // 16x64 per wave
        gemm_bf16_wmma<<<(tiles + 7) / 8, 256, 0, stream>>>(A, lda, W, bias, C, Cbf,
                                                            ldc, M, N, K, act);
    };

    // weights -> bf16
    cvt(in_proj_w,  w_in, 2048 * 512);
    cvt(x_proj_w,   w_xp, 64 * 1024);
    cvt(dt_proj_w,  w_dt, 1024 * 32);
    cvt(out_proj_w, w_op, 512 * 1024);
    cvt(lin1_w,     w_l1, 512 * 512);
    cvt(lin2_w,     w_l2, 512 * 512);

    // 1) audio matrix (bf16 GEMM operand)
    build_avm<<<(64 * 11 * 11 * 512) / 256, 256, 0, stream>>>(src_q, src_v, avm_b);

    // 2) in_proj: (7744,512) @ (512,2048) -> xz f32
    gemm(avm_b, 512, w_in, nullptr, xz, nullptr, 2048, MR, 2048, 512, 0);

    // 3) conv + silu -> f32 + bf16
    conv_silu<<<(704 * 11 * 1024) / 256, 256, 0, stream>>>(xz, conv_w, conv_b, xc, xc_b);

    // 4) x_proj: (7744,1024) @ (1024,64) -> dbl f32 + bf16
    gemm(xc_b, 1024, w_xp, nullptr, dbl, dbl_b, 64, MR, 64, 1024, 0);

    // 5) dt_proj + softplus: (7744,32 strided) @ (32,1024) -> delta f32
    gemm(dbl_b, 64, w_dt, dt_proj_b, delta, nullptr, 1024, MR, 1024, 32, 2);

    // 6) selective scan (+ D skip + silu(z) gate) -> bf16
    mamba_scan<<<dim3(704, 4), 256, 0, stream>>>(xz, xc, dbl, delta, A_log, D_p, yact_b);

    // 7) out_proj: (7744,1024) @ (1024,512) -> out_audio f32
    gemm(yact_b, 1024, w_op, nullptr, oaud, nullptr, 512, MR, 512, 1024, 0);

    // 8) compress (dot over flat groups of 11)
    compress_proj<<<(360448 + 255) / 256, 256, 0, stream>>>(oaud, proj_w, proj_b, af, 360448);

    // 9) src1 + residual + LN1 -> f32 + bf16
    src1_ln1<<<640, 256, 0, stream>>>(src_q, af, ln1_g, ln1_b, qbuf, q_b);

    // 10) FFN
    gemm(q_b,  512, w_l1, lin1_b, nullptr, h1_b, 512, 640, 512, 512, 1);
    gemm(h1_b, 512, w_l2, lin2_b, ffb, nullptr, 512, 640, 512, 512, 0);

    // 11) residual + LN2 -> output
    ln2_final<<<640, 256, 0, stream>>>(qbuf, ffb, ln2_g, ln2_b, out);
}
